// DeeperGCN_17222818857325
// MI455X (gfx1250) — matrix operand order repaired and checked by reading, not compiled, not run on hardware
//
#include <hip/hip_runtime.h>
#include <hip/hip_bf16.h>

// ---------------- problem constants ----------------
constexpr int NN = 100000;
constexpr int EE = 600000;
constexpr int GG = 512;
constexpr int LL = 7;
constexpr int CC = 2;
#define HD 128

typedef __attribute__((ext_vector_type(16))) __bf16 v16bf;
typedef __attribute__((ext_vector_type(8)))  __bf16 v8bf;
typedef __attribute__((ext_vector_type(8)))  float  v8f;

__device__ __forceinline__ unsigned short f2bf(float f) {
  unsigned u = __float_as_uint(f);
  u += 0x7FFFu + ((u >> 16) & 1u);          // round-to-nearest-even
  return (unsigned short)(u >> 16);
}

// ---------------- utility kernels ----------------
__global__ void fill_kernel(float* __restrict__ p, float v, int n) {
  int i = blockIdx.x * blockDim.x + threadIdx.x;
  if (i < n) p[i] = v;
}

__global__ void copy_kernel(float* __restrict__ d, const float* __restrict__ s, int n) {
  int i = blockIdx.x * blockDim.x + threadIdx.x;
  if (i < n) d[i] = s[i];
}

__global__ void init_vn_kernel(float* __restrict__ vn, const float* __restrict__ vne, int n) {
  int i = blockIdx.x * blockDim.x + threadIdx.x;
  if (i < n) vn[i] = vne[i & (HD - 1)];
}

// ---------------- encoders ----------------
// h[n,:] = sum_f atom_emb[f, x[n,f], :] + vn_emb[:]
__global__ void atom_enc_kernel(const int* __restrict__ x, const float* __restrict__ emb,
                                const float* __restrict__ vne, float* __restrict__ h) {
  int tid = blockIdx.x * blockDim.x + threadIdx.x;
  int n = tid >> 5; if (n >= NN) return;
  int c = (tid & 31) * 4;
  float4 acc = *(const float4*)(vne + c);
  #pragma unroll
  for (int f = 0; f < 9; ++f) {
    int v = x[n * 9 + f];
    float4 e = *(const float4*)(emb + ((size_t)f * 64 + v) * HD + c);
    acc.x += e.x; acc.y += e.y; acc.z += e.z; acc.w += e.w;
  }
  *(float4*)(h + (size_t)n * HD + c) = acc;
}

__global__ void bond_enc_kernel(const int* __restrict__ ea, const float* __restrict__ emb,
                                float* __restrict__ ee) {
  int tid = blockIdx.x * blockDim.x + threadIdx.x;
  int e = tid >> 5; if (e >= EE) return;
  int c = (tid & 31) * 4;
  float4 acc = {0.f, 0.f, 0.f, 0.f};
  #pragma unroll
  for (int f = 0; f < 3; ++f) {
    int v = ea[e * 3 + f];
    float4 t = *(const float4*)(emb + ((size_t)f * 8 + v) * HD + c);
    acc.x += t.x; acc.y += t.y; acc.z += t.z; acc.w += t.w;
  }
  *(float4*)(ee + (size_t)e * HD + c) = acc;
}

// ---------------- GENConv edge passes ----------------
__device__ __forceinline__ float4 msg4(const float* __restrict__ h, const float* __restrict__ ee,
                                       int s, int e, int c) {
  float4 hv = *(const float4*)(h + (size_t)s * HD + c);
  float4 ev = *(const float4*)(ee + (size_t)e * HD + c);
  float4 m;
  m.x = fmaxf(hv.x + ev.x, 0.f) + 1e-7f;
  m.y = fmaxf(hv.y + ev.y, 0.f) + 1e-7f;
  m.z = fmaxf(hv.z + ev.z, 0.f) + 1e-7f;
  m.w = fmaxf(hv.w + ev.w, 0.f) + 1e-7f;
  return m;
}

// pass 1: segment max (msg>0 so int-ordered atomicMax on bit pattern is valid; mx init to 0)
__global__ void msg_max_kernel(const float* __restrict__ h, const float* __restrict__ ee,
                               const int* __restrict__ src, const int* __restrict__ dst,
                               int* __restrict__ mx) {
  int tid = blockIdx.x * blockDim.x + threadIdx.x;
  int e = tid >> 5; if (e >= EE) return;
  int c = (tid & 31) * 4;
  float4 m = msg4(h, ee, src[e], e, c);
  int* p = mx + (size_t)dst[e] * HD + c;
  atomicMax(p + 0, __float_as_int(m.x));
  atomicMax(p + 1, __float_as_int(m.y));
  atomicMax(p + 2, __float_as_int(m.z));
  atomicMax(p + 3, __float_as_int(m.w));
}

// pass 2: recompute msg, softmax numer/denom partial sums
__global__ void msg_soft_kernel(const float* __restrict__ h, const float* __restrict__ ee,
                                const int* __restrict__ src, const int* __restrict__ dst,
                                const float* __restrict__ mx,
                                float* __restrict__ den, float* __restrict__ num) {
  int tid = blockIdx.x * blockDim.x + threadIdx.x;
  int e = tid >> 5; if (e >= EE) return;
  int c = (tid & 31) * 4;
  int d = dst[e];
  float4 m = msg4(h, ee, src[e], e, c);
  float4 mv = *(const float4*)(mx + (size_t)d * HD + c);
  float e0 = __expf(m.x - mv.x), e1 = __expf(m.y - mv.y);
  float e2 = __expf(m.z - mv.z), e3 = __expf(m.w - mv.w);
  float* dp = den + (size_t)d * HD + c;
  float* np = num + (size_t)d * HD + c;
  atomicAdd(dp + 0, e0); atomicAdd(dp + 1, e1); atomicAdd(dp + 2, e2); atomicAdd(dp + 3, e3);
  atomicAdd(np + 0, m.x * e0); atomicAdd(np + 1, m.y * e1);
  atomicAdd(np + 2, m.z * e2); atomicAdd(np + 3, m.w * e3);
}

// t = h + num / (den + 1e-16)
__global__ void combine_kernel(const float* __restrict__ h, const float* __restrict__ num,
                               const float* __restrict__ den, float* __restrict__ t) {
  int tid = blockIdx.x * blockDim.x + threadIdx.x;
  int n = tid >> 5; if (n >= NN) return;
  size_t o = (size_t)n * HD + (tid & 31) * 4;
  float4 hv = *(const float4*)(h + o);
  float4 nu = *(const float4*)(num + o);
  float4 de = *(const float4*)(den + o);
  float4 r;
  r.x = hv.x + nu.x / (de.x + 1e-16f);
  r.y = hv.y + nu.y / (de.y + 1e-16f);
  r.z = hv.z + nu.z / (de.z + 1e-16f);
  r.w = hv.w + nu.w / (de.w + 1e-16f);
  *(float4*)(t + o) = r;
}

// ---------------- WMMA GEMM: Y[M,128] = X[M,128] @ W[128,128] + bias (+ res) ----------------
__global__ __launch_bounds__(256) void gemm128_bf16_wmma(
    const float* __restrict__ X, const float* __restrict__ W,
    const float* __restrict__ bias, const float* __restrict__ res,
    float* __restrict__ Y, int M) {
  __shared__ unsigned short As[HD * HD];   // A tile, row-major [m][k], bf16
  __shared__ unsigned short Bs[HD * HD];   // W transposed:  Bs[n*128+k] = W[k*128+n], bf16
  const int t = threadIdx.x;
  const int row0 = blockIdx.x * HD;

  #pragma unroll 4
  for (int i = 0; i < 64; ++i) {
    int idx = i * 256 + t;
    int k = idx >> 7, n = idx & 127;
    Bs[n * HD + k] = f2bf(W[idx]);
  }
  #pragma unroll 4
  for (int i = 0; i < 64; ++i) {
    int idx = i * 256 + t;
    int r = idx >> 7, k = idx & 127;
    int gr = row0 + r;
    As[idx] = f2bf(gr < M ? X[(size_t)gr * HD + k] : 0.f);
  }
  __syncthreads();

  const int wave = t >> 5;
  const int lane = t & 31;
  const int half = lane >> 4;
  const int l16  = lane & 15;

  const v8f vz = {0.f, 0.f, 0.f, 0.f, 0.f, 0.f, 0.f, 0.f};
  v8f acc[8];
  #pragma unroll
  for (int nt = 0; nt < 8; ++nt) acc[nt] = vz;

  union BV { v8bf h[2]; v16bf v; };

  #pragma unroll
  for (int kc = 0; kc < 4; ++kc) {
    // A fragment: lane holds row (wave*16 + l16); elems 0-7: K = kc*32+half*8..+7,
    // elems 8-15: K = kc*32+16+half*8..+7  (ISA 7.12.2, 16-bit A 16x32)
    BV a;
    const int aoff = (wave * 16 + l16) * HD + kc * 32 + half * 8;
    a.h[0] = *(const v8bf*)(As + aoff);
    a.h[1] = *(const v8bf*)(As + aoff + 16);
    #pragma unroll
    for (int nt = 0; nt < 8; ++nt) {
      // B fragment: lane holds column nt*16+l16; elems 0-7: K = kc*32+half*8..+7, elems 8-15: K = kc*32+16+half*8..+7
      BV b;
      const int boff = (nt * 16 + l16) * HD + kc * 32 + half * 8;
      b.h[0] = *(const v8bf*)(Bs + boff);
      b.h[1] = *(const v8bf*)(Bs + boff + 16);
      acc[nt] = __builtin_amdgcn_wmma_f32_16x16x32_bf16(
          false, a.v, false, b.v, (short)0, acc[nt], false, false);
    }
  }

  // C/D layout: VGPR r -> M = r + 8*half, N = l16
  #pragma unroll
  for (int nt = 0; nt < 8; ++nt) {
    const int col = nt * 16 + l16;
    const float bv = bias[col];
    #pragma unroll
    for (int r = 0; r < 8; ++r) {
      const int grow = row0 + wave * 16 + half * 8 + r;
      if (grow < M) {
        float v = acc[nt][r] + bv;
        if (res) v += res[(size_t)grow * HD + col];
        Y[(size_t)grow * HD + col] = v;
      }
    }
  }
}

// ---------------- LayerNorm (+optional ReLU), one wave32 per row ----------------
__global__ __launch_bounds__(256) void ln_kernel(const float* __restrict__ X,
                                                 const float* __restrict__ s,
                                                 const float* __restrict__ b,
                                                 float* __restrict__ Y, int M, int doRelu) {
  int row = blockIdx.x * 8 + (threadIdx.x >> 5);
  if (row >= M) return;
  int lane = threadIdx.x & 31;
  int c = lane * 4;
  float4 v = *(const float4*)(X + (size_t)row * HD + c);
  float sum = v.x + v.y + v.z + v.w;
  #pragma unroll
  for (int o = 16; o; o >>= 1) sum += __shfl_xor(sum, o, 32);
  float mean = sum * (1.f / HD);
  float dx = v.x - mean, dy = v.y - mean, dz = v.z - mean, dw = v.w - mean;
  float vs = dx * dx + dy * dy + dz * dz + dw * dw;
  #pragma unroll
  for (int o = 16; o; o >>= 1) vs += __shfl_xor(vs, o, 32);
  float inv = rsqrtf(vs * (1.f / HD) + 1e-5f);
  float4 sc = *(const float4*)(s + c);
  float4 bb = *(const float4*)(b + c);
  float4 r;
  r.x = dx * inv * sc.x + bb.x;
  r.y = dy * inv * sc.y + bb.y;
  r.z = dz * inv * sc.z + bb.z;
  r.w = dw * inv * sc.w + bb.w;
  if (doRelu) {
    r.x = fmaxf(r.x, 0.f); r.y = fmaxf(r.y, 0.f);
    r.z = fmaxf(r.z, 0.f); r.w = fmaxf(r.w, 0.f);
  }
  *(float4*)(Y + (size_t)row * HD + c) = r;
}

// ---------------- segment sum over batch, and vn broadcast-add ----------------
__global__ void segsum_kernel(const float* __restrict__ h, const int* __restrict__ batch,
                              float* __restrict__ out) {
  int tid = blockIdx.x * blockDim.x + threadIdx.x;
  int n = tid >> 5; if (n >= NN) return;
  int c = (tid & 31) * 4;
  int g = batch[n];
  float4 v = *(const float4*)(h + (size_t)n * HD + c);
  float* p = out + (size_t)g * HD + c;
  atomicAdd(p + 0, v.x); atomicAdd(p + 1, v.y);
  atomicAdd(p + 2, v.z); atomicAdd(p + 3, v.w);
}

__global__ void addvn_kernel(float* __restrict__ h, const float* __restrict__ vn,
                             const int* __restrict__ batch) {
  int tid = blockIdx.x * blockDim.x + threadIdx.x;
  int n = tid >> 5; if (n >= NN) return;
  int c = (tid & 31) * 4;
  size_t o = (size_t)n * HD + c;
  float4 hv = *(const float4*)(h + o);
  float4 vv = *(const float4*)(vn + (size_t)batch[n] * HD + c);
  hv.x += vv.x; hv.y += vv.y; hv.z += vv.z; hv.w += vv.w;
  *(float4*)(h + o) = hv;
}

// ---------------- prediction head: out[G,2] = hg @ pred_W + pred_b ----------------
__global__ void pred_kernel(const float* __restrict__ hg, const float* __restrict__ pW,
                            const float* __restrict__ pb, float* __restrict__ out) {
  int tid = blockIdx.x * blockDim.x + threadIdx.x;
  if (tid >= GG * CC) return;
  int g = tid >> 1, c = tid & 1;
  float s = pb[c];
  #pragma unroll 8
  for (int k = 0; k < HD; ++k) s += hg[(size_t)g * HD + k] * pW[k * CC + c];
  out[tid] = s;
}

// ---------------- host orchestration ----------------
static inline int cdiv(int a, int b) { return (a + b - 1) / b; }

extern "C" void kernel_launch(void* const* d_in, const int* in_sizes, int n_in,
                              void* d_out, int out_size, void* d_ws, size_t ws_size,
                              hipStream_t stream) {
  const int*   x     = (const int*)d_in[0];
  const int*   eidx  = (const int*)d_in[1];
  const int*   eattr = (const int*)d_in[2];
  const int*   batch = (const int*)d_in[3];
  const float* aemb  = (const float*)d_in[4];
  const float* bemb  = (const float*)d_in[5];
  const float* vne   = (const float*)d_in[6];
  const float* gW    = (const float*)d_in[7];
  const float* gb    = (const float*)d_in[8];
  const float* ns    = (const float*)d_in[9];
  const float* nb    = (const float*)d_in[10];
  const float* vW1   = (const float*)d_in[11];
  const float* vb1   = (const float*)d_in[12];
  const float* vls   = (const float*)d_in[13];
  const float* vlb   = (const float*)d_in[14];
  const float* vW2   = (const float*)d_in[15];
  const float* vb2   = (const float*)d_in[16];
  const float* pW    = (const float*)d_in[17];
  const float* pb    = (const float*)d_in[18];
  float* out = (float*)d_out;

  const int* src = eidx;
  const int* dst = eidx + EE;

  const size_t NH = (size_t)NN * HD;
  const size_t EH = (size_t)EE * HD;
  const size_t GH = (size_t)GG * HD;

  float* ws = (float*)d_ws;
  float* hA   = ws;                    // [N,H]
  float* hB   = hA + NH;               // [N,H]
  float* h2   = hB + NH;               // [N,H]
  float* tbuf = h2 + NH;               // [N,H]
  float* mx   = tbuf + NH;             // [N,H]  (also read as int* in pass 1)
  float* den  = mx + NH;               // [N,H]
  float* num  = den + NH;              // [N,H]  (mx,den,num contiguous -> single zero-fill)
  float* ee   = num + NH;              // [E,H]
  float* vn   = ee + EH;               // [G,H]
  float* vtmp = vn + GH;               // [G,H]
  float* y1   = vtmp + GH;             // [G,H]
  float* zb   = y1 + GH;               // [G,H]
  float* hg   = zb + GH;               // [G,H]

  const int TB = 256;
  const int gN32 = cdiv(NN * 32, TB);      // node x 32 lanes
  const int gE32 = cdiv(EE * 32, TB);      // edge x 32 lanes
  const int gLNN = cdiv(NN, 8);
  const int gLNG = cdiv(GG, 8);
  const int gGemmN = cdiv(NN, HD);
  const int gGemmG = cdiv(GG, HD);
  const int gFill3 = cdiv((int)(3 * NH), TB);
  const int gGH = cdiv((int)GH, TB);

  auto genconv = [&](const float* hin, float* hout, const float* W, const float* b,
                     const float* res) {
    fill_kernel<<<gFill3, TB, 0, stream>>>(mx, 0.f, (int)(3 * NH));
    msg_max_kernel<<<gE32, TB, 0, stream>>>(hin, ee, src, dst, (int*)mx);
    msg_soft_kernel<<<gE32, TB, 0, stream>>>(hin, ee, src, dst, mx, den, num);
    combine_kernel<<<gN32, TB, 0, stream>>>(hin, num, den, tbuf);
    gemm128_bf16_wmma<<<gGemmN, TB, 0, stream>>>(tbuf, W, b, res, hout, NN);
  };

  // encoders + virtual-node init
  atom_enc_kernel<<<gN32, TB, 0, stream>>>(x, aemb, vne, hA);
  bond_enc_kernel<<<gE32, TB, 0, stream>>>(eattr, bemb, ee);
  init_vn_kernel<<<gGH, TB, 0, stream>>>(vn, vne, (int)GH);

  // layer 0
  genconv(hA, hB, gW, gb, nullptr);
  float* hcur = hB;
  float* halt = hA;

  for (int l = 1; l < LL; ++l) {
    const size_t HH = (size_t)HD * HD;
    ln_kernel<<<gLNN, TB, 0, stream>>>(hcur, ns + (size_t)(l - 1) * HD,
                                       nb + (size_t)(l - 1) * HD, h2, NN, 1);
    copy_kernel<<<gGH, TB, 0, stream>>>(vtmp, vn, (int)GH);
    segsum_kernel<<<gN32, TB, 0, stream>>>(h2, batch, vtmp);
    gemm128_bf16_wmma<<<gGemmG, TB, 0, stream>>>(vtmp, vW1 + (size_t)(l - 1) * HH,
                                                 vb1 + (size_t)(l - 1) * HD, nullptr, y1, GG);
    ln_kernel<<<gLNG, TB, 0, stream>>>(y1, vls + (size_t)(l - 1) * HD,
                                       vlb + (size_t)(l - 1) * HD, zb, GG, 1);
    gemm128_bf16_wmma<<<gGemmG, TB, 0, stream>>>(zb, vW2 + (size_t)(l - 1) * HH,
                                                 vb2 + (size_t)(l - 1) * HD, nullptr, vn, GG);
    addvn_kernel<<<gN32, TB, 0, stream>>>(h2, vn, batch);
    genconv(h2, halt, gW + (size_t)l * HH, gb + (size_t)l * HD, hcur);
    float* tmp = hcur; hcur = halt; halt = tmp;
  }

  // final LN (no relu), pool, predict
  ln_kernel<<<gLNN, TB, 0, stream>>>(hcur, ns + (size_t)(LL - 1) * HD,
                                     nb + (size_t)(LL - 1) * HD, h2, NN, 0);
  fill_kernel<<<gGH, TB, 0, stream>>>(hg, 0.f, (int)GH);
  segsum_kernel<<<gN32, TB, 0, stream>>>(h2, batch, hg);
  pred_kernel<<<cdiv(GG * CC, TB), TB, 0, stream>>>(hg, pW, pb, out);
}
